// MultiHeadAttention_1614907703611
// MI455X (gfx1250) — matrix operand order repaired and checked by reading, not compiled, not run on hardware
//
#include <hip/hip_runtime.h>

#define DIN   1024
#define DOUTT 1024
#define NH    16
#define DH    64
#define BATCH 4
#define SEQ   2048
#define MTOT  (BATCH * SEQ)   // 8192

typedef _Float16 half_t;
typedef __attribute__((ext_vector_type(16))) _Float16 v16h;
typedef __attribute__((ext_vector_type(8)))  _Float16 v8h;
typedef __attribute__((ext_vector_type(4)))  _Float16 v4h;
typedef __attribute__((ext_vector_type(8)))  float    v8f;

__device__ __forceinline__ v8f wmma16x16x32(v16h a, v16h b, v8f c) {
  return __builtin_amdgcn_wmma_f32_16x16x32_f16(false, a, false, b, (short)0, c,
                                                false, false);
}

// Async global->LDS copy, 16B per lane. VDST VGPR holds the LDS byte offset;
// per ISA 10.2 the low 32 bits of a flat LDS-aperture address ARE the LDS
// offset, so truncating the generic pointer is architecturally correct.
__device__ __forceinline__ void async_copy_b128(void* lds_dst, const void* gsrc) {
  const unsigned int ldsoff = (unsigned int)(unsigned long long)lds_dst;
  asm volatile("global_load_async_to_lds_b128 %0, %1, off"
               :: "v"(ldsoff), "v"(gsrc) : "memory");
}
template <int N>
__device__ __forceinline__ void wait_async() {
  asm volatile("s_wait_asynccnt %0" :: "n"(N) : "memory");
}

// A fragment (16x32 f16): lane m = lane&15, K chunks at (lane>>4)*8 and +16.
__device__ __forceinline__ v16h load_a_frag(const half_t* base, int ld) {
  const int lane = threadIdx.x & 31;
  const int m  = lane & 15;
  const int kb = (lane >> 4) * 8;
  const half_t* p = base + (size_t)m * ld + kb;
  v8h lo = *(const v8h*)(p);
  v8h hi = *(const v8h*)(p + 16);
  v16h r;
#pragma unroll
  for (int i = 0; i < 8; ++i) { r[i] = lo[i]; r[8 + i] = hi[i]; }
  return r;
}

// B fragment (32x16 f16) from an [N][K]-major tile (row n holds K contiguous):
// lane n = lane&15, K chunks at (lane>>4)*8 and +16.
__device__ __forceinline__ v16h load_bt_frag(const half_t* base, int ld) {
  const int lane = threadIdx.x & 31;
  const int n  = lane & 15;
  const int kb = (lane >> 4) * 8;
  const half_t* p = base + (size_t)n * ld + kb;
  v8h lo = *(const v8h*)(p);
  v8h hi = *(const v8h*)(p + 16);
  v16h r;
#pragma unroll
  for (int i = 0; i < 8; ++i) { r[i] = lo[i]; r[8 + i] = hi[i]; }
  return r;
}

__global__ __launch_bounds__(256) void cvt_f32_f16(const float* __restrict__ in,
                                                   half_t* __restrict__ out,
                                                   int n4) {
  int i = blockIdx.x * blockDim.x + threadIdx.x;
  if (i >= n4) return;
  const float4 v = ((const float4*)in)[i];
  v4h o;
  o[0] = (half_t)v.x; o[1] = (half_t)v.y; o[2] = (half_t)v.z; o[3] = (half_t)v.w;
  *(v4h*)(out + (size_t)i * 4) = o;
}

// f32 [K][N] -> f16 [N][K] (transposed), 32x32 tiles via LDS.
__global__ __launch_bounds__(256) void cvt_transpose_f32_f16(
    const float* __restrict__ in, half_t* __restrict__ out) {
  __shared__ half_t t[32][33];
  const int n0 = blockIdx.x * 32, k0 = blockIdx.y * 32;
  const int tx = threadIdx.x & 31, ty = threadIdx.x >> 5;  // 32 x 8
#pragma unroll
  for (int j = 0; j < 32; j += 8)
    t[ty + j][tx] = (half_t)in[(size_t)(k0 + ty + j) * DOUTT + n0 + tx];
  __syncthreads();
#pragma unroll
  for (int j = 0; j < 32; j += 8)
    out[(size_t)(n0 + ty + j) * DIN + k0 + tx] = t[tx][ty + j];
}

// Fused QKV projection. Block tile 128(M) x 64(N), K-slab 32, double-buffered
// async LDS staging; each wave owns a 32x32 register tile (4 WMMAs/slab).
// Weights are pre-transposed [N][K]. Q pre-scaled by 1/sqrt(Dh).
// Q,K stored [B,H,S,Dh]; V stored transposed [B,H,Dh,S].
__global__ __launch_bounds__(256) void qkv_gemm(
    const half_t* __restrict__ xh,
    const half_t* __restrict__ Wqt, const half_t* __restrict__ Wkt,
    const half_t* __restrict__ Wvt,
    const float* __restrict__ bq, const float* __restrict__ bk,
    const float* __restrict__ bv,
    half_t* __restrict__ Qh, half_t* __restrict__ Kh, half_t* __restrict__ Vh) {
  __shared__ __align__(16) half_t As[2][128][40];
  __shared__ __align__(16) half_t Bt[2][64][40];

  const int z = blockIdx.z;
  const half_t* Wt  = (z == 0) ? Wqt : (z == 1) ? Wkt : Wvt;
  const float* bias = (z == 0) ? bq  : (z == 1) ? bk  : bv;
  half_t* Out       = (z == 0) ? Qh  : (z == 1) ? Kh  : Vh;
  const float scale = (z == 0) ? 0.125f : 1.0f;  // 1/sqrt(64)

  const int m0 = blockIdx.x * 128;
  const int n0 = blockIdx.y * 64;
  const int tid  = threadIdx.x;
  const int wave = tid >> 5;
  const int lane = tid & 31;
  const int mi = (wave & 3) * 32;
  const int ni = (wave >> 2) * 32;

  // staging coordinates (16B chunks): A 128x32, B 64x32
  const int ar = tid >> 1, ac = (tid & 1) * 16;
  const int br = tid >> 2, bc = (tid & 3) * 8;

  auto stage = [&](int buf, int k0) {
    async_copy_b128(&As[buf][ar][ac],     &xh[(size_t)(m0 + ar) * DIN + k0 + ac]);
    async_copy_b128(&As[buf][ar][ac + 8], &xh[(size_t)(m0 + ar) * DIN + k0 + ac + 8]);
    async_copy_b128(&Bt[buf][br][bc],     &Wt[(size_t)(n0 + br) * DIN + k0 + bc]);
  };

  v8f acc00 = {}, acc01 = {}, acc10 = {}, acc11 = {};
  const int NS = DIN / 32;
  stage(0, 0);
  for (int i = 0; i < NS; ++i) {
    if (i + 1 < NS) { stage((i + 1) & 1, (i + 1) * 32); wait_async<3>(); }
    else            { wait_async<0>(); }
    __syncthreads();
    const int buf = i & 1;
    const v16h a0 = load_a_frag(&As[buf][mi][0], 40);
    const v16h a1 = load_a_frag(&As[buf][mi + 16][0], 40);
    const v16h b0 = load_bt_frag(&Bt[buf][ni][0], 40);
    const v16h b1 = load_bt_frag(&Bt[buf][ni + 16][0], 40);
    acc00 = wmma16x16x32(a0, b0, acc00);
    acc01 = wmma16x16x32(a0, b1, acc01);
    acc10 = wmma16x16x32(a1, b0, acc10);
    acc11 = wmma16x16x32(a1, b1, acc11);
    __syncthreads();
  }

  const int nl = lane & 15, hi16 = lane >> 4;
#pragma unroll
  for (int rr = 0; rr < 2; ++rr) {
#pragma unroll
    for (int cc = 0; cc < 2; ++cc) {
      const v8f a = (rr == 0) ? (cc == 0 ? acc00 : acc01)
                              : (cc == 0 ? acc10 : acc11);
      const int ng = n0 + ni + cc * 16 + nl;
      const int h = ng >> 6, dh = ng & 63;
      const float bval = bias[ng];
#pragma unroll
      for (int i = 0; i < 8; ++i) {
        const int mg = m0 + mi + rr * 16 + i + 8 * hi16;
        const int bb = mg >> 11, ss = mg & (SEQ - 1);
        const half_t v = (half_t)((a[i] + bval) * scale);
        if (z == 2)  // V transposed: [B,H,Dh,S]
          Out[(((size_t)bb * NH + h) * DH + dh) * SEQ + ss] = v;
        else
          Out[(((size_t)bb * NH + h) * SEQ + ss) * DH + dh] = v;
      }
    }
  }
}

// Flash-attention forward, causal. One block per (b, h, 128-query tile);
// 8 waves, each owns 16 queries x Dh=64. K is [B,H,S,Dh], V is [B,H,Dh,S].
__global__ __launch_bounds__(256) void attn_fwd(const half_t* __restrict__ Qh,
                                                const half_t* __restrict__ Kh,
                                                const half_t* __restrict__ Vh,
                                                half_t* __restrict__ ctxh) {
  __shared__ __align__(16) half_t Ks[32][72];      // [key][d]
  __shared__ __align__(16) half_t Vts[64][40];     // [d][key]
  __shared__ __align__(16) half_t Ps[8][16][40];   // per-wave P scratch [q][k]

  const int bidx = blockIdx.z;
  const int h    = blockIdx.y;
  const int q0   = blockIdx.x * 128;
  const size_t bh = (size_t)bidx * NH + h;
  const half_t* Qb = Qh + bh * SEQ * DH;
  const half_t* Kb = Kh + bh * SEQ * DH;
  const half_t* Vb = Vh + bh * DH * SEQ;   // transposed layout

  const int tid  = threadIdx.x;
  const int wave = tid >> 5;
  const int lane = tid & 31;
  const int nl   = lane & 15;
  const int hi16 = lane >> 4;
  const int qw   = q0 + wave * 16;

  const v16h qa0 = load_a_frag(Qb + (size_t)qw * DH, DH);
  const v16h qa1 = load_a_frag(Qb + (size_t)qw * DH + 32, DH);

  v8f acc[4] = {};
  float rmax[8], rsum[8];
#pragma unroll
  for (int i = 0; i < 8; ++i) { rmax[i] = -1e30f; rsum[i] = 0.f; }

  // staging coordinates: K 32x64 (1 op), Vt 64x32 (1 op)
  const int kr = tid >> 3, kc = (tid & 7) * 8;
  const int vr = tid >> 2, vc = (tid & 3) * 8;

  const int kmax = q0 + 128;
  const int qend = qw + 15;

  for (int kb = 0; kb < kmax; kb += 32) {
    async_copy_b128(&Ks[kr][kc],  &Kb[(size_t)(kb + kr) * DH + kc]);
    async_copy_b128(&Vts[vr][vc], &Vb[(size_t)vr * SEQ + kb + vc]);
    wait_async<0>();
    __syncthreads();

    if (kb <= qend) {  // wave-uniform: EXEC all-ones at every WMMA
      float p[2][8], cand[8];
#pragma unroll
      for (int i = 0; i < 8; ++i) cand[i] = -1e30f;
      v8f sf[2];
#pragma unroll
      for (int ks = 0; ks < 2; ++ks) {
        v8f s = {};
        s = wmma16x16x32(qa0, load_bt_frag(&Ks[ks * 16][0], 72), s);
        s = wmma16x16x32(qa1, load_bt_frag(&Ks[ks * 16][32], 72), s);
        const int kcol = kb + ks * 16 + nl;
#pragma unroll
        for (int i = 0; i < 8; ++i) {
          const int qrow = qw + i + 8 * hi16;
          if (kcol > qrow) s[i] = -1e30f;
          cand[i] = fmaxf(cand[i], s[i]);
        }
        sf[ks] = s;
      }
#pragma unroll
      for (int i = 0; i < 8; ++i) {
        float v = cand[i];
#pragma unroll
        for (int m = 1; m < 16; m <<= 1) v = fmaxf(v, __shfl_xor(v, m, 32));
        const float mnew = fmaxf(rmax[i], v);
        const float f = __expf(rmax[i] - mnew);
        float rs = 0.f;
#pragma unroll
        for (int ks = 0; ks < 2; ++ks) {
          const float sv = sf[ks][i];
          const float pv = (sv > -1e29f) ? __expf(sv - mnew) : 0.f;
          p[ks][i] = pv;
          rs += pv;
        }
#pragma unroll
        for (int m = 1; m < 16; m <<= 1) rs += __shfl_xor(rs, m, 32);
        rsum[i] = rsum[i] * f + rs;
        rmax[i] = mnew;
        acc[0][i] *= f; acc[1][i] *= f; acc[2][i] *= f; acc[3][i] *= f;
      }
#pragma unroll
      for (int ks = 0; ks < 2; ++ks)
#pragma unroll
        for (int i = 0; i < 8; ++i)
          Ps[wave][i + 8 * hi16][ks * 16 + nl] = (half_t)p[ks][i];
      // per-wave LDS is in-order; DScnt wait + compiler barrier suffices
      asm volatile("s_wait_dscnt 0x0" ::: "memory");
      const v16h pa = load_a_frag(&Ps[wave][0][0], 40);
#pragma unroll
      for (int j = 0; j < 4; ++j)
        acc[j] = wmma16x16x32(pa, load_bt_frag(&Vts[j * 16][0], 40), acc[j]);
    }
    __syncthreads();
  }

#pragma unroll
  for (int j = 0; j < 4; ++j) {
#pragma unroll
    for (int i = 0; i < 8; ++i) {
      const int qrow = qw + i + 8 * hi16;
      const int dh = j * 16 + nl;
      const float v = acc[j][i] / rsum[i];
      ctxh[((size_t)bidx * SEQ + qrow) * DOUTT + h * DH + dh] = (half_t)v;
    }
  }
}

// Output projection: ctx[M,1024] @ Wo^T[N][K] + bo -> f32 out. Same tiling as
// qkv_gemm (128x64 block tile, 32x32 per wave, double-buffered async staging).
__global__ __launch_bounds__(256) void out_gemm(const half_t* __restrict__ Ah,
                                                const half_t* __restrict__ Wot,
                                                const float* __restrict__ bo,
                                                float* __restrict__ out) {
  __shared__ __align__(16) half_t As[2][128][40];
  __shared__ __align__(16) half_t Bt[2][64][40];

  const int m0 = blockIdx.x * 128;
  const int n0 = blockIdx.y * 64;
  const int tid  = threadIdx.x;
  const int wave = tid >> 5;
  const int lane = tid & 31;
  const int mi = (wave & 3) * 32;
  const int ni = (wave >> 2) * 32;

  const int ar = tid >> 1, ac = (tid & 1) * 16;
  const int br = tid >> 2, bc = (tid & 3) * 8;

  auto stage = [&](int buf, int k0) {
    async_copy_b128(&As[buf][ar][ac],     &Ah[(size_t)(m0 + ar) * DOUTT + k0 + ac]);
    async_copy_b128(&As[buf][ar][ac + 8], &Ah[(size_t)(m0 + ar) * DOUTT + k0 + ac + 8]);
    async_copy_b128(&Bt[buf][br][bc],     &Wot[(size_t)(n0 + br) * DOUTT + k0 + bc]);
  };

  v8f acc00 = {}, acc01 = {}, acc10 = {}, acc11 = {};
  const int NS = DOUTT / 32;
  stage(0, 0);
  for (int i = 0; i < NS; ++i) {
    if (i + 1 < NS) { stage((i + 1) & 1, (i + 1) * 32); wait_async<3>(); }
    else            { wait_async<0>(); }
    __syncthreads();
    const int buf = i & 1;
    const v16h a0 = load_a_frag(&As[buf][mi][0], 40);
    const v16h a1 = load_a_frag(&As[buf][mi + 16][0], 40);
    const v16h b0 = load_bt_frag(&Bt[buf][ni][0], 40);
    const v16h b1 = load_bt_frag(&Bt[buf][ni + 16][0], 40);
    acc00 = wmma16x16x32(a0, b0, acc00);
    acc01 = wmma16x16x32(a0, b1, acc01);
    acc10 = wmma16x16x32(a1, b0, acc10);
    acc11 = wmma16x16x32(a1, b1, acc11);
    __syncthreads();
  }

  const int nl = lane & 15, hi16 = lane >> 4;
#pragma unroll
  for (int rr = 0; rr < 2; ++rr) {
#pragma unroll
    for (int cc = 0; cc < 2; ++cc) {
      const v8f a = (rr == 0) ? (cc == 0 ? acc00 : acc01)
                              : (cc == 0 ? acc10 : acc11);
      const int ng = n0 + ni + cc * 16 + nl;
      const float bval = bo[ng];
#pragma unroll
      for (int i = 0; i < 8; ++i) {
        const int mg = m0 + mi + rr * 16 + i + 8 * hi16;
        out[(size_t)mg * DOUTT + ng] = a[i] + bval;
      }
    }
  }
}

extern "C" void kernel_launch(void* const* d_in, const int* in_sizes, int n_in,
                              void* d_out, int out_size, void* d_ws,
                              size_t ws_size, hipStream_t stream) {
  (void)in_sizes; (void)n_in; (void)out_size; (void)ws_size;
  const float* x  = (const float*)d_in[0];
  const float* Wq = (const float*)d_in[1];
  const float* bq = (const float*)d_in[2];
  const float* Wk = (const float*)d_in[3];
  const float* bk = (const float*)d_in[4];
  const float* Wv = (const float*)d_in[5];
  const float* bv = (const float*)d_in[6];
  const float* Wo = (const float*)d_in[7];
  const float* bo = (const float*)d_in[8];
  float* out = (float*)d_out;

  // Workspace layout (f16 elements): ~92.3 MB total.
  half_t* ws = (half_t*)d_ws;
  const size_t NX = (size_t)MTOT * DIN;                 // 8,388,608
  const size_t NW = (size_t)DIN * DOUTT;                // 1,048,576
  const size_t NQ = (size_t)BATCH * NH * SEQ * DH;      // 8,388,608
  half_t* xh   = ws;
  half_t* Wqt  = xh + NX;
  half_t* Wkt  = Wqt + NW;
  half_t* Wvt  = Wkt + NW;
  half_t* Wot  = Wvt + NW;
  half_t* Qh   = Wot + NW;
  half_t* Kh   = Qh + NQ;
  half_t* Vh   = Kh + NQ;
  half_t* ctxh = Vh + NQ;

  cvt_f32_f16<<<(int)(NX / 4 / 256), 256, 0, stream>>>(x, xh, (int)(NX / 4));
  cvt_transpose_f32_f16<<<dim3(32, 32), 256, 0, stream>>>(Wq, Wqt);
  cvt_transpose_f32_f16<<<dim3(32, 32), 256, 0, stream>>>(Wk, Wkt);
  cvt_transpose_f32_f16<<<dim3(32, 32), 256, 0, stream>>>(Wv, Wvt);
  cvt_transpose_f32_f16<<<dim3(32, 32), 256, 0, stream>>>(Wo, Wot);

  qkv_gemm<<<dim3(MTOT / 128, DOUTT / 64, 3), 256, 0, stream>>>(
      xh, Wqt, Wkt, Wvt, bq, bk, bv, Qh, Kh, Vh);

  attn_fwd<<<dim3(SEQ / 128, NH, BATCH), 256, 0, stream>>>(Qh, Kh, Vh, ctxh);

  out_gemm<<<dim3(MTOT / 128, DOUTT / 64), 256, 0, stream>>>(ctxh, Wot, bo, out);
}
